// MuseformerDecoderLayer_67439576482208
// MI455X (gfx1250) — compile-verified, matrix operand order checked by reading, not run on hardware
//
#include <hip/hip_runtime.h>

// ---------------- types / constants ----------------
typedef __attribute__((ext_vector_type(16))) __bf16          bf16x16;
typedef __attribute__((ext_vector_type(8)))  float           floatx8;
typedef __attribute__((ext_vector_type(8)))  unsigned short  ushortx8;
typedef __attribute__((ext_vector_type(4)))  unsigned int    uint4v;
typedef __attribute__((ext_vector_type(8)))  int             int8v;
typedef __attribute__((ext_vector_type(4)))  int             int4v;

static constexpr int EMBED    = 1024;
static constexpr int HEADS    = 16;
static constexpr int HEAD_DIM = 64;
static constexpr int FFN      = 4096;
static constexpr int BSZ      = 2;
static constexpr int CHUNK    = 128;
static constexpr int NCHUNK   = 16;
static constexpr int SUM_LEN  = 16;
static constexpr int REG_LEN  = CHUNK * NCHUNK;    // 2048
static constexpr int TOTAL    = SUM_LEN + REG_LEN; // 2064
static constexpr int MTOT     = BSZ * TOTAL;       // 4128

#define DEV __device__ __forceinline__

DEV float bf2f(unsigned short u) {
    unsigned int x = ((unsigned int)u) << 16;
    return __uint_as_float(x);
}
DEV unsigned short f2bf(float f) {
    unsigned int u = __float_as_uint(f);
    u = (u + 0x7FFFu + ((u >> 16) & 1u)) >> 16;   // round-to-nearest-even
    return (unsigned short)u;
}

// ---------------- f32 -> bf16 convert ----------------
__global__ void cvt_kernel(const float* __restrict__ in, unsigned short* __restrict__ out, int n) {
    int i = blockIdx.x * blockDim.x + threadIdx.x;
    if (i < n) out[i] = f2bf(in[i]);
}

// ---------------- LayerNorm (row per block, 256 thr) ----------------
__global__ __launch_bounds__(256)
void ln_kernel(const float* __restrict__ x,
               const float* __restrict__ gs, const float* __restrict__ bs,
               const float* __restrict__ gr, const float* __restrict__ br,
               unsigned short* __restrict__ out) {
    const int row = blockIdx.x;
    const int tl  = row % TOTAL;
    const float* gp = (tl < SUM_LEN) ? gs : gr;
    const float* bp = (tl < SUM_LEN) ? bs : br;
    const int tid = threadIdx.x;

    const float4* xr = reinterpret_cast<const float4*>(x + (size_t)row * EMBED);
    float4 v = xr[tid];
    float s  = v.x + v.y + v.z + v.w;
    float s2 = v.x * v.x + v.y * v.y + v.z * v.z + v.w * v.w;

    __shared__ float r1[256], r2[256];
    r1[tid] = s; r2[tid] = s2;
    __syncthreads();
    for (int ofs = 128; ofs > 0; ofs >>= 1) {
        if (tid < ofs) { r1[tid] += r1[tid + ofs]; r2[tid] += r2[tid + ofs]; }
        __syncthreads();
    }
    float mu   = r1[0] * (1.0f / EMBED);
    float var  = r2[0] * (1.0f / EMBED) - mu * mu;
    float rstd = rsqrtf(var + 1e-5f);

    float4 gv = reinterpret_cast<const float4*>(gp)[tid];
    float4 bv = reinterpret_cast<const float4*>(bp)[tid];
    unsigned short* o4 = out + (size_t)row * EMBED + tid * 4;
    o4[0] = f2bf((v.x - mu) * rstd * gv.x + bv.x);
    o4[1] = f2bf((v.y - mu) * rstd * gv.y + bv.y);
    o4[2] = f2bf((v.z - mu) * rstd * gv.z + bv.z);
    o4[3] = f2bf((v.w - mu) * rstd * gv.w + bv.w);
}

// ---------------- fragment helpers ----------------
// A/B fragment (16x32 bf16): lane%16 = matrix row; lanes 0-15 hold K {0..7,16..23},
// lanes 16-31 hold K {8..15,24..31} (CDNA5 ISA 7.12.2).
struct FragBits { ushortx8 lo, hi; };

DEV bf16x16 lds_frag(const unsigned short* rowptr, int kl, int half) {
    const ushortx8* p = reinterpret_cast<const ushortx8*>(rowptr + kl + half * 8);
    FragBits fb;
    fb.lo = p[0];   // K offsets  +0..7   (relative to half)
    fb.hi = p[2];   // K offsets +16..23  (relative to half)
    return __builtin_bit_cast(bf16x16, fb);
}

// ---------------- TDM: stage a [tileRows x 64] bf16 tile into LDS ----------------
// 2D descriptor per ISA 8.3-8.5: data_size=2B, tile_dim0=64 elems (128B rows),
// pad_enable with interval=32 DWORDs (128B) and amount=4 DWORDs (16B)
// -> LDS row stride 144B (72 shorts), bank-conflict-free b128 fragment reads.
// tensor_dim1 = rows remaining so ragged tails are hardware zero-filled.
static constexpr int LDSROW = 72;   // shorts per staged row (64 data + 8 pad)

DEV void tdm_load_2d(unsigned lds_byte_off, const unsigned short* gptr,
                     unsigned kElems, unsigned rowsRem, unsigned tileRows) {
    unsigned long long ga = (unsigned long long)reinterpret_cast<unsigned long long>(gptr);
    uint4v g0;
    g0.x = 1u;                                               // count=1, user mode
    g0.y = lds_byte_off;                                     // lds_addr
    g0.z = (unsigned)ga;                                     // global_addr[31:0]
    g0.w = (unsigned)((ga >> 32) & 0x1FFFFFFu) | 0x80000000u; // global_addr[56:32] | type=2
    unsigned w0 = (1u << 16) | (1u << 20) | (4u << 22) | (3u << 25); // data_size=2B, pad_en, interval, amount
    unsigned w1 = (kElems & 0xFFFFu) << 16;                  // tensor_dim0[15:0]
    unsigned w2 = (kElems >> 16) | ((rowsRem & 0xFFFFu) << 16); // dim0 hi | tensor_dim1 lo
    unsigned w3 = (rowsRem >> 16) | (64u << 16);             // dim1 hi | tile_dim0=64
    unsigned w4 = tileRows & 0xFFFFu;                        // tile_dim1 (tile_dim2=0)
    unsigned w5 = kElems;                                    // tensor_dim0_stride[31:0]
    int8v g1 = { (int)w0, (int)w1, (int)w2, (int)w3, (int)w4, (int)w5, 0, 0 };
    int4v g2 = { 0, 0, 0, 0 };
    int4v g3 = { 0, 0, 0, 0 };
#if defined(__clang_major__) && (__clang_major__ >= 23)
    int8v gz = { 0, 0, 0, 0, 0, 0, 0, 0 };
    __builtin_amdgcn_tensor_load_to_lds(g0, g1, g2, g3, gz, 0);
#else
    __builtin_amdgcn_tensor_load_to_lds(g0, g1, g2, g3, 0);
#endif
}

// ---------------- TDM-staged WMMA GEMM:  C[M,N] = A[M,K] * B[N,K]^T ----------------
// block = 128 threads (4 waves), tile 128x64, K staged 64 at a time,
// double-buffered LDS (54 KB). Wave grid 2x2; each wave computes 4x2 16x16 tiles
// (8 WMMA per 32-deep k-step vs 6 LDS fragment reads).
static constexpr int BM = 128, BN = 64, BK = 64;
static constexpr int A_TILE = BM * LDSROW;        // shorts
static constexpr int B_TILE = BN * LDSROW;
static constexpr int STAGE  = A_TILE + B_TILE;    // 13824 shorts = 27648 B

template <bool RELU, bool HAS_BIAS, bool HAS_RES, bool OUT_BF16>
__global__ __launch_bounds__(128)
void gemm_tdm(const unsigned short* __restrict__ A,
              const unsigned short* __restrict__ B,
              const float* __restrict__ bias,
              const float* __restrict__ res,
              float* __restrict__ outF,
              unsigned short* __restrict__ outB,
              int M, int N, int K) {
    __shared__ __align__(16) unsigned short lds[2 * STAGE];

    const int lane = threadIdx.x & 31;
    const int wave = threadIdx.x >> 5;
    const int wr   = wave >> 1, wc = wave & 1;
    const int half = lane >> 4, rit = lane & 15;
    const int m0 = blockIdx.y * BM;
    const int n0 = blockIdx.x * BN;

    const unsigned ldsbase = (unsigned)reinterpret_cast<unsigned long long>(&lds[0]);
    const unsigned rowsA = (unsigned)(M - m0);
    const unsigned rowsB = (unsigned)(N - n0);
    const int nstages = K / BK;

    if (wave == 0) {   // prologue: stage 0 into buffer 0
        tdm_load_2d(ldsbase,              A + (size_t)m0 * K, (unsigned)K, rowsA, BM);
        tdm_load_2d(ldsbase + A_TILE * 2, B + (size_t)n0 * K, (unsigned)K, rowsB, BN);
        __builtin_amdgcn_s_wait_tensorcnt(0);
    }
    __syncthreads();

    floatx8 acc[4][2] = {};

    for (int st = 0; st < nstages; ++st) {
        const int cur = st & 1;
        if (wave == 0 && st + 1 < nstages) {   // prefetch next stage via TDM
            const unsigned off = ldsbase + (unsigned)((cur ^ 1) * (STAGE * 2));
            const int k1 = (st + 1) * BK;
            tdm_load_2d(off,              A + (size_t)m0 * K + k1, (unsigned)K, rowsA, BM);
            tdm_load_2d(off + A_TILE * 2, B + (size_t)n0 * K + k1, (unsigned)K, rowsB, BN);
        }
        const unsigned short* Ab = &lds[cur * STAGE];
        const unsigned short* Bb = Ab + A_TILE;
        #pragma unroll
        for (int kk = 0; kk < 2; ++kk) {
            const int kl = kk * 32;
            bf16x16 fa[4], fb[2];
            #pragma unroll
            for (int mt = 0; mt < 4; ++mt)
                fa[mt] = lds_frag(Ab + (wr * 64 + mt * 16 + rit) * LDSROW, kl, half);
            #pragma unroll
            for (int nt = 0; nt < 2; ++nt)
                fb[nt] = lds_frag(Bb + (wc * 32 + nt * 16 + rit) * LDSROW, kl, half);
            #pragma unroll
            for (int mt = 0; mt < 4; ++mt)
                #pragma unroll
                for (int nt = 0; nt < 2; ++nt)
                    acc[mt][nt] = __builtin_amdgcn_wmma_f32_16x16x32_bf16(
                        false, fa[mt], false, fb[nt], (short)0, acc[mt][nt], false, false);
        }
        if (wave == 0 && st + 1 < nstages) __builtin_amdgcn_s_wait_tensorcnt(0);
        __syncthreads();
    }

    #pragma unroll
    for (int mt = 0; mt < 4; ++mt) {
        #pragma unroll
        for (int nt = 0; nt < 2; ++nt) {
            #pragma unroll
            for (int r = 0; r < 8; ++r) {
                // C/D layout: lanes 0-15 hold M=r, lanes 16-31 hold M=r+8, N=lane%16
                int row = m0 + wr * 64 + mt * 16 + r + half * 8;
                int col = n0 + wc * 32 + nt * 16 + rit;
                if (row < M) {
                    float vv = acc[mt][nt][r];
                    if (HAS_BIAS) vv += bias[col];
                    if (HAS_RES)  vv += res[(size_t)row * N + col];
                    if (RELU)     vv = fmaxf(vv, 0.0f);
                    if (OUT_BF16) outB[(size_t)row * N + col] = f2bf(vv);
                    else          outF[(size_t)row * N + col] = vv;
                }
            }
        }
    }
}

// ---------------- Block-sparse attention ----------------
// grid = (HEADS, NCHUNK+1, BSZ), block = 128.
// group 0: 16 sum queries (keys: sums 0..s causal + all regs of chunk s)
// group g>=1: 128 reg queries of chunk c=g-1
//             (keys: sums 0..c-1 + chunk-local causal regs) via LDS-staged K/V.
DEV float dotq(const float* q, const unsigned short* k) {
    float s = 0.0f;
    #pragma unroll
    for (int d = 0; d < HEAD_DIM; ++d) s = fmaf(q[d], bf2f(k[d]), s);
    return s;
}

__global__ __launch_bounds__(128)
void attn_kernel(const unsigned short* __restrict__ Q,
                 const unsigned short* __restrict__ K,
                 const unsigned short* __restrict__ V,
                 unsigned short* __restrict__ ctx) {
    const int h  = blockIdx.x;
    const int g  = blockIdx.y;
    const int b  = blockIdx.z;
    const int tb = b * TOTAL;
    const int hoff = h * HEAD_DIM;
    const float scale = 0.125f;  // 1/sqrt(64)
    const int tid = threadIdx.x;

    __shared__ __align__(16) unsigned short Ks[(SUM_LEN + CHUNK) * HEAD_DIM];
    __shared__ __align__(16) unsigned short Vs[(SUM_LEN + CHUNK) * HEAD_DIM];

    if (g == 0) {
        if (tid < SUM_LEN) {
            const int s = tid;
            float q[HEAD_DIM];
            const unsigned short* qp = Q + (size_t)(tb + s) * EMBED + hoff;
            #pragma unroll
            for (int d = 0; d < HEAD_DIM; ++d) q[d] = bf2f(qp[d]);

            float m = -3e38f, l = 0.0f;
            for (int j = 0; j <= s; ++j) {
                float sc = dotq(q, K + (size_t)(tb + j) * EMBED + hoff) * scale;
                float mn = fmaxf(m, sc);
                l = l * __expf(m - mn) + __expf(sc - mn);
                m = mn;
            }
            for (int u = 0; u < CHUNK; ++u) {
                int kt = SUM_LEN + s * CHUNK + u;
                float sc = dotq(q, K + (size_t)(tb + kt) * EMBED + hoff) * scale;
                float mn = fmaxf(m, sc);
                l = l * __expf(m - mn) + __expf(sc - mn);
                m = mn;
            }
            float acc[HEAD_DIM];
            #pragma unroll
            for (int d = 0; d < HEAD_DIM; ++d) acc[d] = 0.0f;
            for (int j = 0; j <= s; ++j) {
                const unsigned short* kp = K + (size_t)(tb + j) * EMBED + hoff;
                const unsigned short* vp = V + (size_t)(tb + j) * EMBED + hoff;
                float p = __expf(dotq(q, kp) * scale - m);
                #pragma unroll
                for (int d = 0; d < HEAD_DIM; ++d) acc[d] = fmaf(p, bf2f(vp[d]), acc[d]);
            }
            for (int u = 0; u < CHUNK; ++u) {
                int kt = SUM_LEN + s * CHUNK + u;
                const unsigned short* kp = K + (size_t)(tb + kt) * EMBED + hoff;
                const unsigned short* vp = V + (size_t)(tb + kt) * EMBED + hoff;
                float p = __expf(dotq(q, kp) * scale - m);
                #pragma unroll
                for (int d = 0; d < HEAD_DIM; ++d) acc[d] = fmaf(p, bf2f(vp[d]), acc[d]);
            }
            float inv = 1.0f / l;
            unsigned short* op = ctx + (size_t)(tb + s) * EMBED + hoff;
            #pragma unroll
            for (int d = 0; d < HEAD_DIM; ++d) op[d] = f2bf(acc[d] * inv);
        }
        return;
    }

    const int c = g - 1;
    // stage: rows [0,16) = sum tokens, rows [16,144) = chunk-c reg tokens
    for (int r = tid; r < SUM_LEN + CHUNK; r += 128) {
        int kt = (r < SUM_LEN) ? r : (SUM_LEN + c * CHUNK + (r - SUM_LEN));
        const ushortx8* sk = reinterpret_cast<const ushortx8*>(K + (size_t)(tb + kt) * EMBED + hoff);
        const ushortx8* sv = reinterpret_cast<const ushortx8*>(V + (size_t)(tb + kt) * EMBED + hoff);
        ushortx8* dk = reinterpret_cast<ushortx8*>(&Ks[r * HEAD_DIM]);
        ushortx8* dv = reinterpret_cast<ushortx8*>(&Vs[r * HEAD_DIM]);
        #pragma unroll
        for (int j = 0; j < HEAD_DIM / 8; ++j) { dk[j] = sk[j]; dv[j] = sv[j]; }
    }
    __syncthreads();

    const int t = SUM_LEN + c * CHUNK + tid;   // this thread's query token
    float q[HEAD_DIM];
    const unsigned short* qp = Q + (size_t)(tb + t) * EMBED + hoff;
    #pragma unroll
    for (int d = 0; d < HEAD_DIM; ++d) q[d] = bf2f(qp[d]);

    float m = -3e38f, l = 0.0f;
    for (int j = 0; j < c; ++j) {   // previous-chunk summaries (strict <)
        float sc = dotq(q, &Ks[j * HEAD_DIM]) * scale;
        float mn = fmaxf(m, sc);
        l = l * __expf(m - mn) + __expf(sc - mn);
        m = mn;
    }
    for (int k = 0; k <= tid; ++k) { // chunk-local causal
        float sc = dotq(q, &Ks[(SUM_LEN + k) * HEAD_DIM]) * scale;
        float mn = fmaxf(m, sc);
        l = l * __expf(m - mn) + __expf(sc - mn);
        m = mn;
    }
    float acc[HEAD_DIM];
    #pragma unroll
    for (int d = 0; d < HEAD_DIM; ++d) acc[d] = 0.0f;
    for (int j = 0; j < c; ++j) {
        float p = __expf(dotq(q, &Ks[j * HEAD_DIM]) * scale - m);
        const unsigned short* vp = &Vs[j * HEAD_DIM];
        #pragma unroll
        for (int d = 0; d < HEAD_DIM; ++d) acc[d] = fmaf(p, bf2f(vp[d]), acc[d]);
    }
    for (int k = 0; k <= tid; ++k) {
        float p = __expf(dotq(q, &Ks[(SUM_LEN + k) * HEAD_DIM]) * scale - m);
        const unsigned short* vp = &Vs[(SUM_LEN + k) * HEAD_DIM];
        #pragma unroll
        for (int d = 0; d < HEAD_DIM; ++d) acc[d] = fmaf(p, bf2f(vp[d]), acc[d]);
    }
    float inv = 1.0f / l;
    unsigned short* op = ctx + (size_t)(tb + t) * EMBED + hoff;
    #pragma unroll
    for (int d = 0; d < HEAD_DIM; ++d) op[d] = f2bf(acc[d] * inv);
}

// ---------------- host orchestration ----------------
extern "C" void kernel_launch(void* const* d_in, const int* in_sizes, int n_in,
                              void* d_out, int out_size, void* d_ws, size_t ws_size,
                              hipStream_t stream) {
    (void)in_sizes; (void)n_in; (void)out_size; (void)ws_size;

    const float* x       = (const float*)d_in[0];
    const float* w_q     = (const float*)d_in[1];
    const float* w_k     = (const float*)d_in[2];
    const float* w_v     = (const float*)d_in[3];
    const float* w_o     = (const float*)d_in[4];
    const float* ln1_sg  = (const float*)d_in[5];
    const float* ln1_sb  = (const float*)d_in[6];
    const float* ln1_rg  = (const float*)d_in[7];
    const float* ln1_rb  = (const float*)d_in[8];
    const float* fc1s_w  = (const float*)d_in[9];
    const float* fc1s_b  = (const float*)d_in[10];
    const float* fc2s_w  = (const float*)d_in[11];
    const float* fc2s_b  = (const float*)d_in[12];
    const float* fc1r_w  = (const float*)d_in[13];
    const float* fc1r_b  = (const float*)d_in[14];
    const float* fc2r_w  = (const float*)d_in[15];
    const float* fc2r_b  = (const float*)d_in[16];
    const float* ln2_sg  = (const float*)d_in[17];
    const float* ln2_sb  = (const float*)d_in[18];
    const float* ln2_rg  = (const float*)d_in[19];
    const float* ln2_rb  = (const float*)d_in[20];
    float* outF = (float*)d_out;

    // workspace carve-up (bf16 elements)
    unsigned short* wsp = (unsigned short*)d_ws;
    auto alloc = [&](size_t n) { unsigned short* p = wsp; wsp += n; return p; };
    unsigned short* h1   = alloc((size_t)MTOT * EMBED);   // reused as h2 later
    unsigned short* wqb  = alloc((size_t)EMBED * EMBED);
    unsigned short* wkb  = alloc((size_t)EMBED * EMBED);
    unsigned short* wvb  = alloc((size_t)EMBED * EMBED);
    unsigned short* wob  = alloc((size_t)EMBED * EMBED);
    unsigned short* f1sb = alloc((size_t)FFN * EMBED);
    unsigned short* f2sb = alloc((size_t)EMBED * FFN);
    unsigned short* f1rb = alloc((size_t)FFN * EMBED);
    unsigned short* f2rb = alloc((size_t)EMBED * FFN);
    unsigned short* Qb   = alloc((size_t)MTOT * EMBED);   // Q,K,V,ctx contiguous:
    unsigned short* Kb   = alloc((size_t)MTOT * EMBED);   // aliased later as the
    unsigned short* Vb   = alloc((size_t)MTOT * EMBED);   // FFN hidden buffer
    unsigned short* ctxb = alloc((size_t)MTOT * EMBED);
    (void)ctxb;
    unsigned short* ffn1 = Qb;   // 4*MTOT*EMBED == MTOT*FFN exactly
    unsigned short* h2   = h1;   // h1 dead after QKV projections

    // --- weights -> bf16 (L2-resident afterwards) ---
    const int nW = EMBED * EMBED, nF = FFN * EMBED;
    cvt_kernel<<<(nW + 255) / 256, 256, 0, stream>>>(w_q, wqb, nW);
    cvt_kernel<<<(nW + 255) / 256, 256, 0, stream>>>(w_k, wkb, nW);
    cvt_kernel<<<(nW + 255) / 256, 256, 0, stream>>>(w_v, wvb, nW);
    cvt_kernel<<<(nW + 255) / 256, 256, 0, stream>>>(w_o, wob, nW);
    cvt_kernel<<<(nF + 255) / 256, 256, 0, stream>>>(fc1s_w, f1sb, nF);
    cvt_kernel<<<(nF + 255) / 256, 256, 0, stream>>>(fc2s_w, f2sb, nF);
    cvt_kernel<<<(nF + 255) / 256, 256, 0, stream>>>(fc1r_w, f1rb, nF);
    cvt_kernel<<<(nF + 255) / 256, 256, 0, stream>>>(fc2r_w, f2rb, nF);

    // --- LN1 ---
    ln_kernel<<<MTOT, 256, 0, stream>>>(x, ln1_sg, ln1_sb, ln1_rg, ln1_rb, h1);

    // --- QKV projections ---
    dim3 gQ(EMBED / BN, (MTOT + BM - 1) / BM);
    gemm_tdm<false, false, false, true><<<gQ, 128, 0, stream>>>(h1, wqb, nullptr, nullptr, nullptr, Qb, MTOT, EMBED, EMBED);
    gemm_tdm<false, false, false, true><<<gQ, 128, 0, stream>>>(h1, wkb, nullptr, nullptr, nullptr, Kb, MTOT, EMBED, EMBED);
    gemm_tdm<false, false, false, true><<<gQ, 128, 0, stream>>>(h1, wvb, nullptr, nullptr, nullptr, Vb, MTOT, EMBED, EMBED);

    // --- block-sparse attention ---
    attn_kernel<<<dim3(HEADS, NCHUNK + 1, BSZ), 128, 0, stream>>>(Qb, Kb, Vb, ctxb);

    // --- output projection + residual -> d_out (f32) ---
    gemm_tdm<false, false, true, false><<<gQ, 128, 0, stream>>>(ctxb, wob, nullptr, x, outF, nullptr, MTOT, EMBED, EMBED);

    // --- LN2 ---
    ln_kernel<<<MTOT, 256, 0, stream>>>(outF, ln2_sg, ln2_sb, ln2_rg, ln2_rb, h2);

    // --- FFN per segment (sum rows use sum weights, reg rows use reg weights) ---
    struct Seg { int base; int m; bool sum; };
    const Seg segs[4] = {
        { 0,               SUM_LEN, true  },
        { SUM_LEN,         REG_LEN, false },
        { TOTAL,           SUM_LEN, true  },
        { TOTAL + SUM_LEN, REG_LEN, false },
    };
    for (int i = 0; i < 4; ++i) {
        const Seg sg = segs[i];
        const unsigned short* f1w = sg.sum ? f1sb : f1rb;
        const float*          f1b = sg.sum ? fc1s_b : fc1r_b;
        const unsigned short* f2w = sg.sum ? f2sb : f2rb;
        const float*          f2b = sg.sum ? fc2s_b : fc2r_b;
        dim3 ga(FFN / BN, (sg.m + BM - 1) / BM);
        gemm_tdm<true, true, false, true><<<ga, 128, 0, stream>>>(
            h2 + (size_t)sg.base * EMBED, f1w, f1b, nullptr, nullptr,
            ffn1 + (size_t)sg.base * FFN, sg.m, FFN, EMBED);
        dim3 gb(EMBED / BN, (sg.m + BM - 1) / BM);
        gemm_tdm<false, true, true, false><<<gb, 128, 0, stream>>>(
            ffn1 + (size_t)sg.base * FFN, f2w, f2b,
            outF + (size_t)sg.base * EMBED, outF + (size_t)sg.base * EMBED, nullptr,
            sg.m, EMBED, FFN);
    }
}
